// Qwen3Attention_36661840839153
// MI455X (gfx1250) — matrix operand order repaired and checked by reading, not compiled, not run on hardware
//
#include <hip/hip_runtime.h>

#define HIDDEN 4096
#define NUM_Q 32
#define NUM_KV 8
#define HEAD_DIM 128

typedef unsigned short u16;
typedef __bf16 bf16_t;
typedef bf16_t v16bf __attribute__((ext_vector_type(16)));
typedef float v8f __attribute__((ext_vector_type(8)));

union Frag { v16bf v; uint4 u[2]; };

__device__ __forceinline__ u16 f2bf(float f) {
  unsigned int u = __float_as_uint(f);
  u += 0x7FFFu + ((u >> 16) & 1u);   // round-to-nearest-even
  return (u16)(u >> 16);
}

// ---------------------------------------------------------------------------
// CDNA5 async global->LDS copy (no data VGPRs, tracked by ASYNCcnt)
// Builtin signature (from compiler diagnostic): param0 is
//   int __attribute__((vector_size(16))) __device__ *   (global int4*)
// ---------------------------------------------------------------------------
typedef int i32x4_vs __attribute__((__vector_size__(16)));

#if __has_builtin(__builtin_amdgcn_global_load_async_to_lds_b128)
#define HAVE_ASYNC_LDS 1
__device__ __forceinline__ void async_cp16(const u16* g, u16* l) {
  __builtin_amdgcn_global_load_async_to_lds_b128(
      (__attribute__((address_space(1))) i32x4_vs*)g,
      (__attribute__((address_space(3))) i32x4_vs*)l, 0, 0);
}
#else
#define HAVE_ASYNC_LDS 0
#endif

__device__ __forceinline__ void wait_async0() {
#if __has_builtin(__builtin_amdgcn_s_wait_asynccnt)
  __builtin_amdgcn_s_wait_asynccnt(0);
#else
  asm volatile("s_wait_asynccnt 0x0" ::: "memory");
#endif
}

// ---------------------------------------------------------------------------
// f32 -> bf16 conversion (grid-stride), row-major preserved (used for hidden)
// ---------------------------------------------------------------------------
__global__ void to_bf16(const float* __restrict__ in, u16* __restrict__ out, size_t n) {
  size_t i = (size_t)blockIdx.x * blockDim.x + threadIdx.x;
  size_t stride = (size_t)gridDim.x * blockDim.x;
  for (; i < n; i += stride) out[i] = f2bf(in[i]);
}

// ---------------------------------------------------------------------------
// f32 [K][N] -> bf16 [N][K] transposed conversion (one-time per weight).
// ---------------------------------------------------------------------------
__global__ __launch_bounds__(256)
void transpose_to_bf16(const float* __restrict__ in, u16* __restrict__ out,
                       int Kdim, int Ndim) {
  __shared__ u16 tile[32][33];
  const int tx = threadIdx.x & 31;
  const int ty = threadIdx.x >> 5;  // 0..7
  const int nb = blockIdx.x * 32, kb = blockIdx.y * 32;
#pragma unroll
  for (int j = 0; j < 4; ++j)
    tile[ty + j * 8][tx] = f2bf(in[(size_t)(kb + ty + j * 8) * Ndim + nb + tx]);
  __syncthreads();
#pragma unroll
  for (int j = 0; j < 4; ++j)
    out[(size_t)(nb + ty + j * 8) * Kdim + kb + tx] = tile[tx][ty + j * 8];
}

// ---------------------------------------------------------------------------
// GEMM: C[M,N] (f32) = A[M,K] (bf16 row-major) * Bt[N,K] (bf16, pre-transposed)
// Block: 128 thr (4 waves), tile 128x128, K-step 32, double-buffered LDS.
// Wave w (2x2): 64x64 sub-tile => 16 v_wmma per K-step.
// ---------------------------------------------------------------------------
__global__ __launch_bounds__(128, 1)
void gemm_bf16(const u16* __restrict__ A, const u16* __restrict__ Bt,
               float* __restrict__ C, int M, int N, int K) {
  __shared__ __attribute__((aligned(16))) u16 Alds[2][128 * 40];  // [m][k]
  __shared__ __attribute__((aligned(16))) u16 Blds[2][128 * 40];  // [n][k]

  const int t = threadIdx.x;
  const int lane = t & 31, w = t >> 5;
  const int half = lane >> 4, r = lane & 15;
  const int m0 = blockIdx.y * 128, n0 = blockIdx.x * 128;
  const int wm = (w >> 1) * 64, wn = (w & 1) * 64;

  // per-thread staging addresses (4 rows apart by 32, fixed k-segment)
  const int srow = t >> 2;
  const int sks = (t & 3) * 8;
  const u16* pa[4];
  const u16* pb[4];
  int soff[4];
#pragma unroll
  for (int i = 0; i < 4; ++i) {
    int row = srow + i * 32;
    pa[i] = A + (size_t)(m0 + row) * K + sks;
    pb[i] = Bt + (size_t)(n0 + row) * K + sks;
    soff[i] = row * 40 + sks;
  }

  const v8f vzero = {0.f, 0.f, 0.f, 0.f, 0.f, 0.f, 0.f, 0.f};
  v8f acc[4][4];
#pragma unroll
  for (int mi = 0; mi < 4; ++mi)
#pragma unroll
    for (int ni = 0; ni < 4; ++ni) acc[mi][ni] = vzero;

  const int ak0 = half * 8, ak1 = 16 + half * 8;   // A frag K chunks
  const int bk0 = half * 8, bk1 = 16 + half * 8;   // B frag K chunks

  // prologue: stage K-step 0 into buffer 0
#if HAVE_ASYNC_LDS
#pragma unroll
  for (int i = 0; i < 4; ++i) {
    async_cp16(pa[i], &Alds[0][soff[i]]);
    async_cp16(pb[i], &Blds[0][soff[i]]);
  }
  wait_async0();
#else
  {
    uint4 ta[4], tb[4];
#pragma unroll
    for (int i = 0; i < 4; ++i) {
      ta[i] = *(const uint4*)(pa[i]);
      tb[i] = *(const uint4*)(pb[i]);
    }
#pragma unroll
    for (int i = 0; i < 4; ++i) {
      *(uint4*)(&Alds[0][soff[i]]) = ta[i];
      *(uint4*)(&Blds[0][soff[i]]) = tb[i];
    }
  }
#endif
  __syncthreads();

  int buf = 0;
  for (int k0 = 0; k0 < K; k0 += 32) {
    const bool more = (k0 + 32) < K;
#if HAVE_ASYNC_LDS
    if (more) {  // async-stage next tile directly into alternate LDS buffer
#pragma unroll
      for (int i = 0; i < 4; ++i) {
        async_cp16(pa[i] + k0 + 32, &Alds[buf ^ 1][soff[i]]);
        async_cp16(pb[i] + k0 + 32, &Blds[buf ^ 1][soff[i]]);
      }
      if (k0 + 64 < K) {
        __builtin_prefetch(pa[0] + k0 + 64, 0, 1);
        __builtin_prefetch(pb[0] + k0 + 64, 0, 1);
      }
    }
#else
    uint4 ta[4], tb[4];
    if (more) {
#pragma unroll
      for (int i = 0; i < 4; ++i) {
        ta[i] = *(const uint4*)(pa[i] + k0 + 32);
        tb[i] = *(const uint4*)(pb[i] + k0 + 32);
      }
      if (k0 + 64 < K) {
        __builtin_prefetch(pa[0] + k0 + 64, 0, 1);
        __builtin_prefetch(pb[0] + k0 + 64, 0, 1);
      }
    }
#endif

    const u16* Ab = &Alds[buf][0];
    const u16* Bb = &Blds[buf][0];
    Frag a[4], b[4];
#pragma unroll
    for (int mi = 0; mi < 4; ++mi) {
      const u16* base = Ab + (wm + mi * 16 + r) * 40;
      a[mi].u[0] = *(const uint4*)(base + ak0);
      a[mi].u[1] = *(const uint4*)(base + ak1);
    }
#pragma unroll
    for (int ni = 0; ni < 4; ++ni) {
      const u16* base = Bb + (wn + ni * 16 + r) * 40;
      b[ni].u[0] = *(const uint4*)(base + bk0);
      b[ni].u[1] = *(const uint4*)(base + bk1);
    }
#pragma unroll
    for (int mi = 0; mi < 4; ++mi)
#pragma unroll
      for (int ni = 0; ni < 4; ++ni)
        acc[mi][ni] = __builtin_amdgcn_wmma_f32_16x16x32_bf16(
            false, a[mi].v, false, b[ni].v, (short)0, acc[mi][ni], false, false);

#if HAVE_ASYNC_LDS
    if (more) wait_async0();
#else
    if (more) {
#pragma unroll
      for (int i = 0; i < 4; ++i) {
        *(uint4*)(&Alds[buf ^ 1][soff[i]]) = ta[i];
        *(uint4*)(&Blds[buf ^ 1][soff[i]]) = tb[i];
      }
    }
#endif
    __syncthreads();
    buf ^= 1;
  }

#pragma unroll
  for (int mi = 0; mi < 4; ++mi)
#pragma unroll
    for (int ni = 0; ni < 4; ++ni)
#pragma unroll
      for (int i = 0; i < 8; ++i) {
        int row = m0 + wm + mi * 16 + half * 8 + i;
        int col = n0 + wn + ni * 16 + r;
        C[(size_t)row * N + col] = acc[mi][ni][i];
      }
}

// ---------------------------------------------------------------------------
// RMSNorm + RoPE + bf16 pack. One wave32 per (token, head-slot).
// ---------------------------------------------------------------------------
__global__ __launch_bounds__(256)
void normrope(const float* __restrict__ qf, const float* __restrict__ kf,
              const float* __restrict__ vf, const float* __restrict__ qw,
              const float* __restrict__ kw, const int* __restrict__ pos,
              u16* __restrict__ qb, u16* __restrict__ kb, u16* __restrict__ vtb,
              int n_tok) {
  const int t = threadIdx.x;
  const int lane = t & 31, w = t >> 5;
  const int gw = blockIdx.x * 8 + w;
  const int tok = gw / 48;
  const int hh = gw - tok * 48;
  if (tok >= n_tok) return;

  if (hh >= 40) {  // V: convert + transpose to [kv][d][tok]
    const int h = hh - 40;
    const float* x = vf + ((size_t)tok * NUM_KV + h) * HEAD_DIM;
#pragma unroll
    for (int j = 0; j < 4; ++j) {
      int d = lane + j * 32;
      vtb[((size_t)h * HEAD_DIM + d) * n_tok + tok] = f2bf(x[d]);
    }
    return;
  }

  const float* x;
  const float* nw;
  u16* out;
  if (hh < 32) {
    x = qf + ((size_t)tok * NUM_Q + hh) * HEAD_DIM;
    nw = qw;
    out = qb + ((size_t)tok * NUM_Q + hh) * HEAD_DIM;
  } else {
    int h = hh - 32;
    x = kf + ((size_t)tok * NUM_KV + h) * HEAD_DIM;
    nw = kw;
    out = kb + ((size_t)h * n_tok + tok) * HEAD_DIM;
  }

  float x0 = x[lane], x1 = x[lane + 32], x2 = x[lane + 64], x3 = x[lane + 96];
  float ss = x0 * x0 + x1 * x1 + x2 * x2 + x3 * x3;
#pragma unroll
  for (int o = 16; o > 0; o >>= 1) ss += __shfl_xor(ss, o, 32);
  float inv = rsqrtf(ss * (1.0f / 128.0f) + 1e-6f);
  x0 *= inv * nw[lane];
  x1 *= inv * nw[lane + 32];
  x2 *= inv * nw[lane + 64];
  x3 *= inv * nw[lane + 96];

  const float p = (float)pos[tok];
  const float kfac = 13.815510557964274f / 64.0f;  // ln(1e6)/64
  float f0 = __expf(-(float)lane * kfac);
  float f1 = __expf(-(float)(lane + 32) * kfac);
  float s0, c0, s1, c1;
  __sincosf(p * f0, &s0, &c0);
  __sincosf(p * f1, &s1, &c1);
  float o0 = x0 * c0 - x2 * s0;
  float o2 = x2 * c0 + x0 * s0;
  float o1 = x1 * c1 - x3 * s1;
  float o3 = x3 * c1 + x1 * s1;
  out[lane] = f2bf(o0);
  out[lane + 32] = f2bf(o1);
  out[lane + 64] = f2bf(o2);
  out[lane + 96] = f2bf(o3);
}

// ---------------------------------------------------------------------------
// Flash attention. Block = (kv-head, 32 q rows), 8 waves:
// wave = (q-head in group [w>>1], 16-row half [w&1]). Shares LDS K / V^T tiles.
// ---------------------------------------------------------------------------
__global__ __launch_bounds__(256)
void attn_kernel(const u16* __restrict__ Q, const u16* __restrict__ Kb,
                 const u16* __restrict__ Vt, u16* __restrict__ Ob, int n_tok) {
  __shared__ __attribute__((aligned(16))) u16 Klds[64 * 136];   // [key][d]
  __shared__ __attribute__((aligned(16))) u16 Vlds[128 * 72];   // [d][key]
  __shared__ __attribute__((aligned(16))) float Sw[8][16 * 64]; // per-wave scores
  __shared__ __attribute__((aligned(16))) u16 Pl[8][16 * 72];   // per-wave probs
  __shared__ float Fl[8][16];                                   // per-wave row factors

  const int t = threadIdx.x;
  const int lane = t & 31, w = t >> 5;
  const int half = lane >> 4, r = lane & 15;
  const int qblk = blockIdx.x, hkv = blockIdx.y;
  const int h = hkv * 4 + (w >> 1);
  const int q0 = qblk * 32 + (w & 1) * 16;

  float* sw = &Sw[w][0];
  u16* pl = &Pl[w][0];
  float* fl = &Fl[w][0];

  // per-thread staging base pointers / LDS offsets
  const u16* kg[4];
  const u16* vg[4];
  int koff[4], voff[4];
#pragma unroll
  for (int i = 0; i < 4; ++i) {
    int s = t + i * 256;
    kg[i] = Kb + ((size_t)hkv * n_tok + (s >> 4)) * HEAD_DIM + (s & 15) * 8;
    koff[i] = (s >> 4) * 136 + (s & 15) * 8;
    vg[i] = Vt + ((size_t)hkv * HEAD_DIM + (s >> 3)) * n_tok + (s & 7) * 8;
    voff[i] = (s >> 3) * 72 + (s & 7) * 8;
  }

  // Q fragments (16 rows x 128 dims), held in registers
  Frag qfr[4];
  {
    const int tok = q0 + r;
    const u16* qbase = Q + ((size_t)tok * NUM_Q + h) * HEAD_DIM;
#pragma unroll
    for (int ks = 0; ks < 4; ++ks) {
      qfr[ks].u[0] = *(const uint4*)(qbase + ks * 32 + half * 8);
      qfr[ks].u[1] = *(const uint4*)(qbase + ks * 32 + 16 + half * 8);
    }
  }

  const v8f vzero = {0.f, 0.f, 0.f, 0.f, 0.f, 0.f, 0.f, 0.f};
  v8f acc[8];
#pragma unroll
  for (int dn = 0; dn < 8; ++dn) acc[dn] = vzero;

  float mrow = -1e30f, lrow = 0.f;
  const float scale = 0.08838834764831845f;  // 1/sqrt(128)
  const int kmax = qblk * 32 + 32;
  const int ntile = (kmax + 63) >> 6;

  for (int kt = 0; kt < ntile; ++kt) {
    const int key0 = kt * 64;
    // stage K (64x128) and V^T (128x64)
#if HAVE_ASYNC_LDS
#pragma unroll
    for (int i = 0; i < 4; ++i) {
      async_cp16(kg[i] + (size_t)key0 * HEAD_DIM, Klds + koff[i]);
      async_cp16(vg[i] + key0, Vlds + voff[i]);
    }
#else
    {
      uint4 tk[4], tv[4];
#pragma unroll
      for (int i = 0; i < 4; ++i) {
        tk[i] = *(const uint4*)(kg[i] + (size_t)key0 * HEAD_DIM);
        tv[i] = *(const uint4*)(vg[i] + key0);
      }
#pragma unroll
      for (int i = 0; i < 4; ++i) {
        *(uint4*)(Klds + koff[i]) = tk[i];
        *(uint4*)(Vlds + voff[i]) = tv[i];
      }
    }
#endif
    if (kt + 1 < ntile) {  // prefetch next tile
      __builtin_prefetch(kg[0] + (size_t)(key0 + 64) * HEAD_DIM, 0, 1);
      __builtin_prefetch(vg[0] + key0 + 64, 0, 1);
    }
#if HAVE_ASYNC_LDS
    wait_async0();
#endif
    __syncthreads();

    // scores: (16 q rows) x (64 keys)
    v8f sc[4];
#pragma unroll
    for (int nt = 0; nt < 4; ++nt) sc[nt] = vzero;
#pragma unroll
    for (int nt = 0; nt < 4; ++nt)
#pragma unroll
      for (int ks = 0; ks < 4; ++ks) {
        Frag bk;
        const u16* kbase = Klds + (nt * 16 + r) * 136 + ks * 32;
        bk.u[0] = *(const uint4*)(kbase + half * 8);
        bk.u[1] = *(const uint4*)(kbase + 16 + half * 8);
        sc[nt] = __builtin_amdgcn_wmma_f32_16x16x32_bf16(
            false, qfr[ks].v, false, bk.v, (short)0, sc[nt], false, false);
      }

    // dump scores to per-wave LDS
#pragma unroll
    for (int nt = 0; nt < 4; ++nt)
#pragma unroll
      for (int i = 0; i < 8; ++i)
        sw[(i + half * 8) * 64 + nt * 16 + r] = sc[nt][i];

    // online softmax: 2 lanes per row (lane r and r+16), 32 cols each
    {
      const int row = r;
      const int tokq = q0 + row;
      const int cbase = half * 32;
      float mx = -1e30f;
#pragma unroll
      for (int j = 0; j < 32; ++j) {
        float v = sw[row * 64 + cbase + j] * scale;
        if (key0 + cbase + j > tokq) v = -1e30f;
        mx = fmaxf(mx, v);
      }
      mx = fmaxf(mx, __shfl_xor(mx, 16, 32));
      mx = fmaxf(mx, mrow);
      float corr = __expf(mrow - mx);
      float sum = 0.f;
#pragma unroll
      for (int j = 0; j < 32; ++j) {
        int col = cbase + j;
        float v = sw[row * 64 + col] * scale;
        float pj = (key0 + col > tokq) ? 0.f : __expf(v - mx);
        sum += pj;
        pl[row * 72 + col] = f2bf(pj);
      }
      sum += __shfl_xor(sum, 16, 32);
      lrow = lrow * corr + sum;
      mrow = mx;
      if (lane < 16) fl[row] = corr;
    }

    // rescale running accumulators by per-row correction
    float fr[8];
#pragma unroll
    for (int i = 0; i < 8; ++i) fr[i] = fl[i + half * 8];
#pragma unroll
    for (int dn = 0; dn < 8; ++dn)
#pragma unroll
      for (int i = 0; i < 8; ++i) acc[dn][i] *= fr[i];

    // out += P (16x64) @ V (64x128)
#pragma unroll
    for (int dn = 0; dn < 8; ++dn)
#pragma unroll
      for (int ks2 = 0; ks2 < 2; ++ks2) {
        Frag ap, bv;
        const u16* pb = pl + r * 72 + ks2 * 32;
        ap.u[0] = *(const uint4*)(pb + half * 8);
        ap.u[1] = *(const uint4*)(pb + 16 + half * 8);
        const u16* vb = Vlds + (dn * 16 + r) * 72 + ks2 * 32;
        bv.u[0] = *(const uint4*)(vb + half * 8);
        bv.u[1] = *(const uint4*)(vb + 16 + half * 8);
        acc[dn] = __builtin_amdgcn_wmma_f32_16x16x32_bf16(
            false, ap.v, false, bv.v, (short)0, acc[dn], false, false);
      }
    __syncthreads();
  }

  // final 1/l normalization + store bf16 [tok][h*128 + d]
  if (lane < 16) fl[lane] = 1.0f / lrow;
  float fr[8];
#pragma unroll
  for (int i = 0; i < 8; ++i) fr[i] = fl[i + half * 8];
#pragma unroll
  for (int dn = 0; dn < 8; ++dn)
#pragma unroll
    for (int i = 0; i < 8; ++i) {
      int tok = q0 + i + half * 8;
      int d = dn * 16 + r;
      Ob[((size_t)tok * NUM_Q + h) * HEAD_DIM + d] = f2bf(acc[dn][i] * fr[i]);
    }
}

// ---------------------------------------------------------------------------
extern "C" void kernel_launch(void* const* d_in, const int* in_sizes, int n_in,
                              void* d_out, int out_size, void* d_ws, size_t ws_size,
                              hipStream_t stream) {
  const float* hidden = (const float*)d_in[0];
  const int* positions = (const int*)d_in[1];
  const float* Wq = (const float*)d_in[2];
  const float* Wk = (const float*)d_in[3];
  const float* Wv = (const float*)d_in[4];
  const float* Wo = (const float*)d_in[5];
  const float* qnw = (const float*)d_in[6];
  const float* knw = (const float*)d_in[7];

  const size_t nt = (size_t)in_sizes[0] / HIDDEN;  // 2048
  const size_t QD = (size_t)NUM_Q * HEAD_DIM;      // 4096
  const size_t KD = (size_t)NUM_KV * HEAD_DIM;     // 1024

  auto al = [](size_t x) { return (x + 255) & ~(size_t)255; };
  char* p = (char*)d_ws;
  u16* hid_b = (u16*)p;  p += al(nt * HIDDEN * 2);
  u16* WqT_b = (u16*)p;  p += al(HIDDEN * QD * 2);  // [QD][HIDDEN]
  u16* WkT_b = (u16*)p;  p += al(HIDDEN * KD * 2);  // [KD][HIDDEN]
  u16* WvT_b = (u16*)p;  p += al(HIDDEN * KD * 2);  // [KD][HIDDEN]
  u16* WoT_b = (u16*)p;  p += al(QD * HIDDEN * 2);  // [HIDDEN][QD]
  float* q_f = (float*)p; p += al(nt * QD * 4);
  float* k_f = (float*)p; p += al(nt * KD * 4);
  float* v_f = (float*)p; p += al(nt * KD * 4);
  u16* q_b   = (u16*)p;  p += al(nt * QD * 2);
  u16* k_b   = (u16*)p;  p += al(nt * KD * 2);
  u16* vT_b  = (u16*)p;  p += al(nt * KD * 2);
  u16* att_b = (u16*)q_f;  // alias: q_f32 is dead by the time attention writes

  // 1) convert hidden (row-major) and weights (transposed to [N][K]) to bf16
  to_bf16<<<1024, 256, 0, stream>>>(hidden, hid_b, nt * HIDDEN);
  transpose_to_bf16<<<dim3(QD / 32, HIDDEN / 32), 256, 0, stream>>>(Wq, WqT_b,
                                                                    HIDDEN, (int)QD);
  transpose_to_bf16<<<dim3(KD / 32, HIDDEN / 32), 256, 0, stream>>>(Wk, WkT_b,
                                                                    HIDDEN, (int)KD);
  transpose_to_bf16<<<dim3(KD / 32, HIDDEN / 32), 256, 0, stream>>>(Wv, WvT_b,
                                                                    HIDDEN, (int)KD);
  transpose_to_bf16<<<dim3(HIDDEN / 32, QD / 32), 256, 0, stream>>>(Wo, WoT_b,
                                                                    (int)QD, HIDDEN);

  // 2) QKV projections
  gemm_bf16<<<dim3(QD / 128, nt / 128), 128, 0, stream>>>(hid_b, WqT_b, q_f,
                                                          (int)nt, (int)QD, HIDDEN);
  gemm_bf16<<<dim3(KD / 128, nt / 128), 128, 0, stream>>>(hid_b, WkT_b, k_f,
                                                          (int)nt, (int)KD, HIDDEN);
  gemm_bf16<<<dim3(KD / 128, nt / 128), 128, 0, stream>>>(hid_b, WvT_b, v_f,
                                                          (int)nt, (int)KD, HIDDEN);

  // 3) RMSNorm + RoPE + bf16 pack (q tok-major, k [kv][tok][d], v transposed)
  normrope<<<(unsigned)(nt * 48 / 8), 256, 0, stream>>>(q_f, k_f, v_f, qnw, knw,
                                                        positions, q_b, k_b, vT_b,
                                                        (int)nt);

  // 4) flash attention (writes bf16 into att_b, aliasing dead q_f)
  attn_kernel<<<dim3(nt / 32, NUM_KV), 256, 0, stream>>>(q_b, k_b, vT_b, att_b,
                                                         (int)nt);

  // 5) output projection -> f32 d_out
  gemm_bf16<<<dim3(HIDDEN / 128, nt / 128), 128, 0, stream>>>(
      att_b, WoT_b, (float*)d_out, (int)nt, HIDDEN, (int)QD);
}
